// Encoder_10977936409136
// MI455X (gfx1250) — matrix operand order repaired and checked by reading, not compiled, not run on hardware
//
#include <hip/hip_runtime.h>

// GRU (Keras reset_after=True): B=64, T=512, D=256, H=1024
#define Bz 64
#define Tz 512
#define Dz 256
#define Hz 1024
#define H3 3072

typedef __bf16 bf16;
typedef __attribute__((ext_vector_type(16))) __bf16 v16bf;
typedef __attribute__((ext_vector_type(8)))  float  v8f;

// ---------- fragment loaders (match CDNA5 16-bit WMMA VGPR layouts) ----------
// A 16x32 bf16, row-major source with row stride `stride` (elements).
__device__ __forceinline__ v16bf load_a_frag(const bf16* __restrict__ a, int stride, int kbase) {
    int l = threadIdx.x & 31;
    const bf16* row = a + (size_t)(l & 15) * stride + kbase;
    int hi = (l >> 4) << 3;
    union { v16bf v; uint4 u[2]; } t;
    t.u[0] = *(const uint4*)(row + hi);
    t.u[1] = *(const uint4*)(row + 16 + hi);
    return t.v;
}

// B 32x16 bf16 from a [16 cols x K] row-major panel (each original column contiguous).
// lane l: col = l&15, K = kbase + (l>>4)*8 .. +7 and kbase + 16 + (l>>4)*8 .. +7
__device__ __forceinline__ v16bf load_b_frag(const bf16* p0, int K, int kbase) {
    int l = threadIdx.x & 31;
    const bf16* p = p0 + (size_t)(l & 15) * K + kbase + ((l >> 4) << 3);
    union { v16bf v; uint4 u[2]; } t;
    t.u[0] = *(const uint4*)p;
    t.u[1] = *(const uint4*)(p + 16);
    return t.v;
}

__device__ __forceinline__ v8f wmma_bf16(v16bf a, v16bf b, v8f c) {
    return __builtin_amdgcn_wmma_f32_16x16x32_bf16(false, a, false, b, (short)0, c, false, false);
}

__device__ __forceinline__ float sigmoidf_(float x) { return 1.0f / (1.0f + __expf(-x)); }

// ---------- prep kernels ----------
__global__ void k_prep_wt(const float* __restrict__ W, bf16* __restrict__ WT) {
    int i = blockIdx.x * blockDim.x + threadIdx.x;          // over 3H*D
    if (i >= H3 * Dz) return;
    int n = i / Dz, k = i % Dz;
    WT[i] = (bf16)W[(size_t)k * H3 + n];
}
__global__ void k_prep_ut(const float* __restrict__ U, bf16* __restrict__ UT) {
    int i = blockIdx.x * blockDim.x + threadIdx.x;          // over 3H*H
    if (i >= H3 * Hz) return;
    int n = i / Hz, k = i % Hz;
    UT[i] = (bf16)U[(size_t)k * H3 + n];
}
__global__ void k_prep_x(const float* __restrict__ x, bf16* __restrict__ xb, int n) {
    int i = blockIdx.x * blockDim.x + threadIdx.x;
    if (i < n) xb[i] = (bf16)x[i];
}
__global__ void k_prep_h(const float* __restrict__ h0, float* __restrict__ hf, bf16* __restrict__ hb) {
    int i = blockIdx.x * blockDim.x + threadIdx.x;
    if (i < Bz * Hz) { float v = h0[i]; hf[i] = v; hb[i] = (bf16)v; }
}
__global__ void k_copy_state(const float* __restrict__ hf, float* __restrict__ st) {
    int i = blockIdx.x * blockDim.x + threadIdx.x;
    if (i < Bz * Hz) st[i] = hf[i];
}

// ---------- fused GRU step with LDS-staged B panels ----------
// Block = 8 waves = {4 batch tiles} x {2 hidden tiles}. The block's B panel
// (3 gates x 32 cols x 1024 K of U, + 3 x 32 x 256 of W) = 240 KB lives in the
// CDNA5 320KB-per-WGP LDS, staged once per step, then all WMMAs read B via ds.
__global__ __launch_bounds__(256)
void k_gru_step(const bf16* __restrict__ xbf, const bf16* __restrict__ WT,
                const bf16* __restrict__ UT,
                const float* __restrict__ bi, const float* __restrict__ br,
                const float* __restrict__ hin_f, const bf16* __restrict__ hin_b,
                float* __restrict__ hout_f, bf16* __restrict__ hout_b,
                float* __restrict__ out, int t) {
    __shared__ bf16 uB[3 * 32 * Hz];   // 192 KB: gate g, local col c(0..31), k
    __shared__ bf16 wB[3 * 32 * Dz];   //  48 KB

    int n0 = blockIdx.x << 5;          // 32 hidden cols per block
    int tid = threadIdx.x;

    // ---- cooperative stage: three contiguous 64KB row-blocks of UT, 16KB of WT
    {
        uint4* du = (uint4*)uB;        // 3 * 4096 chunks of 16B
        for (int i = tid; i < 3 * 4096; i += 256) {
            int g = i >> 12, c = i & 4095;
            du[i] = *((const uint4*)(UT + (size_t)(g * Hz + n0) * Hz) + c);
        }
        uint4* dw = (uint4*)wB;        // 3 * 1024 chunks
        for (int i = tid; i < 3 * 1024; i += 256) {
            int g = i >> 10, c = i & 1023;
            dw[i] = *((const uint4*)(WT + (size_t)(g * Hz + n0) * Dz) + c);
        }
    }
    __syncthreads();

    int wv    = tid >> 5;              // 0..7
    int mtile = wv & 3;                // batch tile
    int lnt   = wv >> 2;               // local hidden tile (0/1)
    int m0    = mtile << 4;
    int lc0   = lnt << 4;              // local col base within the 32-col panel

    v8f az = {}, ar = {}, axh = {}, arh = {};

    // recurrent GEMM: h[64,H] @ U[H,3H], K = 1024, B from LDS
    const bf16* Ah  = hin_b + (size_t)m0 * Hz;
    const bf16* uBz = &uB[(size_t)(0 * 32 + lc0) * Hz];
    const bf16* uBr = &uB[(size_t)(1 * 32 + lc0) * Hz];
    const bf16* uBh = &uB[(size_t)(2 * 32 + lc0) * Hz];
    for (int k = 0; k < Hz; k += 32) {
        v16bf a  = load_a_frag(Ah, Hz, k);
        v16bf b0 = load_b_frag(uBz, Hz, k);
        v16bf b1 = load_b_frag(uBr, Hz, k);
        v16bf b2 = load_b_frag(uBh, Hz, k);
        az  = wmma_bf16(a, b0, az);
        ar  = wmma_bf16(a, b1, ar);
        arh = wmma_bf16(a, b2, arh);
    }

    // input GEMM: x_t[64,D] @ W[D,3H], K = 256, B from LDS
    const bf16* Ax  = xbf + (size_t)m0 * Tz * Dz + (size_t)t * Dz;
    const bf16* wBz = &wB[(size_t)(0 * 32 + lc0) * Dz];
    const bf16* wBr = &wB[(size_t)(1 * 32 + lc0) * Dz];
    const bf16* wBh = &wB[(size_t)(2 * 32 + lc0) * Dz];
    for (int k = 0; k < Dz; k += 32) {
        v16bf a  = load_a_frag(Ax, Tz * Dz, k);
        v16bf b0 = load_b_frag(wBz, Dz, k);
        v16bf b1 = load_b_frag(wBr, Dz, k);
        v16bf b2 = load_b_frag(wBh, Dz, k);
        az  = wmma_bf16(a, b0, az);
        ar  = wmma_bf16(a, b1, ar);
        axh = wmma_bf16(a, b2, axh);
    }

    // gate epilogue. C/D layout: VGPR j -> row m = j + (lane>>4)*8, col = lane&15
    int l = tid & 31;
    int n = n0 + lc0 + (l & 15);
    int rofs = (l >> 4) << 3;
    float cz = bi[n] + br[n];
    float cr = bi[Hz + n] + br[Hz + n];
    float ch = bi[2 * Hz + n];
    float dh = br[2 * Hz + n];

    union { v8f v; float e[8]; } uz, ur, uxh, urh;
    uz.v = az; ur.v = ar; uxh.v = axh; urh.v = arh;

#pragma unroll
    for (int j = 0; j < 8; ++j) {
        int b = m0 + j + rofs;
        float z  = sigmoidf_(uz.e[j] + cz);
        float r  = sigmoidf_(ur.e[j] + cr);
        float hh = tanhf(uxh.e[j] + ch + r * (urh.e[j] + dh));
        float hp = hin_f[(size_t)b * Hz + n];
        float hn = z * hp + (1.0f - z) * hh;
        out[((size_t)b * Tz + t) * Hz + n] = hn;
        hout_f[(size_t)b * Hz + n] = hn;
        hout_b[(size_t)b * Hz + n] = (bf16)hn;
    }
}

// ---------- launch ----------
extern "C" void kernel_launch(void* const* d_in, const int* in_sizes, int n_in,
                              void* d_out, int out_size, void* d_ws, size_t ws_size,
                              hipStream_t stream) {
    const float* x      = (const float*)d_in[0];
    const float* hidden = (const float*)d_in[1];
    const float* W      = (const float*)d_in[2];
    const float* U      = (const float*)d_in[3];
    const float* bi     = (const float*)d_in[4];
    const float* br     = (const float*)d_in[5];
    float* out   = (float*)d_out;                      // [B,T,H] then [B,H]
    float* state = out + (size_t)Bz * Tz * Hz;

    char* ws = (char*)d_ws;
    size_t o = 0;
    bf16* WT = (bf16*)(ws + o); o += (size_t)H3 * Dz * 2;          // 1.5 MB
    bf16* UT = (bf16*)(ws + o); o += (size_t)H3 * Hz * 2;          // 6 MB
    bf16* xb = (bf16*)(ws + o); o += (size_t)Bz * Tz * Dz * 2;     // 16 MB
    float* hf0 = (float*)(ws + o); o += (size_t)Bz * Hz * 4;
    float* hf1 = (float*)(ws + o); o += (size_t)Bz * Hz * 4;
    bf16* hb0 = (bf16*)(ws + o); o += (size_t)Bz * Hz * 2;
    bf16* hb1 = (bf16*)(ws + o); o += (size_t)Bz * Hz * 2;

    const int TPB = 256;
    k_prep_wt<<<(H3 * Dz + TPB - 1) / TPB, TPB, 0, stream>>>(W, WT);
    k_prep_ut<<<(H3 * Hz + TPB - 1) / TPB, TPB, 0, stream>>>(U, UT);
    int nx = Bz * Tz * Dz;
    k_prep_x<<<(nx + TPB - 1) / TPB, TPB, 0, stream>>>(x, xb, nx);
    k_prep_h<<<(Bz * Hz + TPB - 1) / TPB, TPB, 0, stream>>>(hidden, hf0, hb0);

    for (int t = 0; t < Tz; ++t) {
        const float* hin_f = (t & 1) ? hf1 : hf0;
        const bf16*  hin_b = (t & 1) ? hb1 : hb0;
        float* hout_f = (t & 1) ? hf0 : hf1;
        bf16*  hout_b = (t & 1) ? hb0 : hb1;
        k_gru_step<<<32, 256, 0, stream>>>(xb, WT, UT, bi, br,
                                           hin_f, hin_b, hout_f, hout_b, out, t);
    }
    // after 512 steps (even), final h lives in buffer 0
    k_copy_state<<<(Bz * Hz + TPB - 1) / TPB, TPB, 0, stream>>>(hf0, state);
}
